// Model_76982993813840
// MI455X (gfx1250) — matrix operand order repaired and checked by reading, not compiled, not run on hardware
//
#include <hip/hip_runtime.h>
#include <math.h>

// ---------------------------------------------------------------------------
// Mamba time-series forecaster for MI455X (gfx1250, wave32).
// GEMMs: v_wmma_f32_16x16x32_bf16, fully unrolled K (templated), 16x32 tile
//        per wave (A-fragment reuse across two B tiles -> 2 WMMAs per K-step).
// Scan:  1 thread per (b, d_inner), state in registers, B/C staged in LDS.
// ---------------------------------------------------------------------------

typedef __attribute__((ext_vector_type(16))) __bf16 v16bf;
typedef __attribute__((ext_vector_type(8)))  float  v8f;

#define BATCH   32
#define LSEQ    1024
#define MROWS   (BATCH * LSEQ)      // 32768 rows, multiple of 16
#define DMODEL  128
#define DINNER  256
#define DSTATE  16
#define DTRANK  8
#define DCONV   4
#define NLAYER  2
#define ENCIN   7
#define MARKF   4
#define COUT    7

// ------------------------------- embed -------------------------------------
__global__ void embed_kernel(const float* __restrict__ x_enc,
                             const float* __restrict__ x_mark,
                             const float* __restrict__ token_W,   // (128,7,3)
                             const float* __restrict__ temporal_W,// (128,4)
                             float* __restrict__ h) {
    int row = blockIdx.x;            // b*L + l
    int d   = threadIdx.x;           // 0..127
    int b = row / LSEQ, l = row % LSEQ;
    float acc = 0.f;
#pragma unroll
    for (int k = 0; k < 3; ++k) {
        int i = l + k - 1;
        i = (i < 0) ? (LSEQ - 1) : ((i >= LSEQ) ? 0 : i);
        const float* xp = x_enc + ((size_t)b * LSEQ + i) * ENCIN;
#pragma unroll
        for (int c = 0; c < ENCIN; ++c)
            acc += xp[c] * token_W[(d * ENCIN + c) * 3 + k];
    }
    const float* mk = x_mark + ((size_t)b * LSEQ + l) * MARKF;
#pragma unroll
    for (int f = 0; f < MARKF; ++f) acc += mk[f] * temporal_W[d * MARKF + f];
    const float LOG1E4_OVER_D = 9.210340371976184f / (float)DMODEL;
    float freq = __expf(-(float)(d & ~1) * LOG1E4_OVER_D);
    float ang  = (float)l * freq;
    acc += (d & 1) ? __cosf(ang) : __sinf(ang);
    h[(size_t)row * DMODEL + d] = acc;
}

// ------------------------------ layernorm ----------------------------------
__global__ void ln_kernel(const float* __restrict__ hidden,
                          const float* resid_in,     // may be null
                          float* resid_out,          // may be null
                          const float* __restrict__ w,
                          const float* __restrict__ b,
                          float* __restrict__ out) {
    __shared__ float red[DMODEL];
    int row = blockIdx.x, d = threadIdx.x;
    float v = hidden[(size_t)row * DMODEL + d];
    if (resid_in)  v += resid_in[(size_t)row * DMODEL + d];
    if (resid_out) resid_out[(size_t)row * DMODEL + d] = v;
    red[d] = v; __syncthreads();
    for (int s = DMODEL / 2; s > 0; s >>= 1) {
        if (d < s) red[d] += red[d + s];
        __syncthreads();
    }
    float mu = red[0] * (1.f / DMODEL);
    __syncthreads();
    float c = v - mu;
    red[d] = c * c; __syncthreads();
    for (int s = DMODEL / 2; s > 0; s >>= 1) {
        if (d < s) red[d] += red[d + s];
        __syncthreads();
    }
    float var = red[0] * (1.f / DMODEL);
    out[(size_t)row * DMODEL + d] = c * rsqrtf(var + 1e-5f) * w[d] + b[d];
}

// ------------------------- WMMA bf16 GEMM ----------------------------------
// C[M,N] = A[M,K] @ W[N,K]^T ; one 16x32 output tile per wave (2 accums),
// compile-time K -> fully unrolled chain of v_wmma_f32_16x16x32_bf16.
// A frag (16x32 bf16): lane<16 -> row=lane, K 0..15;
//                      lane>=16 -> row=lane-16, K 16..31.
// B frag (32x16 bf16): lane = N column (mod 16); lanes 0-15 hold K 0..15,
//                      lanes 16-31 hold K 16..31.
template <int K>
__global__ void __launch_bounds__(256)
gemm_bf16_wmma(const float* __restrict__ A, const float* __restrict__ W,
               float* __restrict__ C, int N, int tilesN2) {
    int wave = threadIdx.x >> 5;
    int lane = threadIdx.x & 31;
    int tile = blockIdx.x * 8 + wave;        // 8 waves per block
    int tm = tile / tilesN2;
    int tn = tile - tm * tilesN2;            // covers 32 output columns
    if (tm >= (MROWS / 16)) return;          // wave-uniform: EXEC stays all-1s
    int  m_base = tm * 16;
    bool hi   = (lane & 16) != 0;
    int  arow = m_base + (lane & 15);
    int  n0   = tn * 32 + (lane & 15);
    int  n1   = n0 + 16;
    bool v0 = (n0 < N), v1 = (n1 < N);
    const float* pa  = A + (size_t)arow * K + (hi ? 16 : 0);
    const float* pb0 = W + (size_t)n0 * K + (hi ? 16 : 0);
    const float* pb1 = W + (size_t)n1 * K + (hi ? 16 : 0);
    v8f acc0 = {}, acc1 = {};
#pragma unroll
    for (int k0 = 0; k0 < K; k0 += 32) {
        v16bf a, b0, b1;
#pragma unroll
        for (int i = 0; i < 8; ++i) {
            a[i]     = (__bf16)pa[k0 + i];
            a[8 + i] = (__bf16)pa[k0 + 8 + i];
        }
        if (v0) {
#pragma unroll
            for (int i = 0; i < 16; ++i) b0[i] = (__bf16)pb0[k0 + i];
        } else {
#pragma unroll
            for (int i = 0; i < 16; ++i) b0[i] = (__bf16)0.0f;
        }
        if (v1) {
#pragma unroll
            for (int i = 0; i < 16; ++i) b1[i] = (__bf16)pb1[k0 + i];
        } else {
#pragma unroll
            for (int i = 0; i < 16; ++i) b1[i] = (__bf16)0.0f;
        }
        acc0 = __builtin_amdgcn_wmma_f32_16x16x32_bf16(
                   false, a, false, b0, (short)0, acc0, false, false);
        acc1 = __builtin_amdgcn_wmma_f32_16x16x32_bf16(
                   false, a, false, b1, (short)0, acc1, false, false);
    }
    int mrow = m_base + (hi ? 8 : 0);
    if (v0) {
#pragma unroll
        for (int r = 0; r < 8; ++r) C[(size_t)(mrow + r) * N + n0] = acc0[r];
    }
    if (v1) {
#pragma unroll
        for (int r = 0; r < 8; ++r) C[(size_t)(mrow + r) * N + n1] = acc1[r];
    }
}

// --------------------- depthwise causal conv + SiLU ------------------------
__global__ void conv_silu_kernel(const float* __restrict__ xz,   // (M,512)
                                 const float* __restrict__ cW,   // (256,4)
                                 const float* __restrict__ cb,
                                 float* __restrict__ u) {
    int idx = blockIdx.x * blockDim.x + threadIdx.x;
    if (idx >= MROWS * DINNER) return;
    int d   = idx % DINNER;
    int row = idx / DINNER;
    int l   = row % LSEQ;
    float acc = cb[d];
#pragma unroll
    for (int w = 0; w < DCONV; ++w) {
        int j = l + w - (DCONV - 1);
        if (j >= 0)
            acc += xz[(size_t)(row + w - (DCONV - 1)) * 512 + d] * cW[d * DCONV + w];
    }
    u[(size_t)row * DINNER + d] = acc / (1.f + __expf(-acc));
}

// -------------------------- dt = softplus(proj) ----------------------------
__global__ void dt_kernel(const float* __restrict__ dbc,   // (M,40), cols 0..7
                          const float* __restrict__ dtW,   // (256,8)
                          const float* __restrict__ dtb,   // (256)
                          float* __restrict__ dt) {
    int idx = blockIdx.x * blockDim.x + threadIdx.x;
    if (idx >= MROWS * DINNER) return;
    int d   = idx % DINNER;
    int row = idx / DINNER;
    const float* r = dbc + (size_t)row * 40;
    float acc = dtb[d];
#pragma unroll
    for (int j = 0; j < DTRANK; ++j) acc += r[j] * dtW[d * DTRANK + j];
    dt[idx] = (acc > 20.f) ? acc : log1pf(__expf(acc));
}

// --------------------------- selective scan --------------------------------
__global__ void __launch_bounds__(DINNER)
scan_kernel(const float* __restrict__ dbc,   // (M,40): B=8..23, C=24..39
            const float* __restrict__ u,     // (M,256)
            const float* __restrict__ xz,    // (M,512): z = cols 256..511
            const float* __restrict__ A_log, // (256,16)
            const float* __restrict__ Dp,    // (256)
            float* dty) {                    // (M,256) dt in, y out (aliased)
    __shared__ float sB[DSTATE], sC[DSTATE];
    int b = blockIdx.x, d = threadIdx.x;
    float A[DSTATE];
#pragma unroll
    for (int s = 0; s < DSTATE; ++s) A[s] = -__expf(A_log[d * DSTATE + s]);
    float Dl = Dp[d];
    float h[DSTATE];
#pragma unroll
    for (int s = 0; s < DSTATE; ++s) h[s] = 0.f;

    for (int l = 0; l < LSEQ; ++l) {
        size_t row = (size_t)b * LSEQ + l;
        if (d < DSTATE)            sB[d]          = dbc[row * 40 + 8 + d];
        else if (d < 2 * DSTATE)   sC[d - DSTATE] = dbc[row * 40 + 8 + d];
        __syncthreads();
        float dtv = dty[row * DINNER + d];
        float uv  = u[row * DINNER + d];
        float du  = dtv * uv;
        float yv  = 0.f;
#pragma unroll
        for (int s = 0; s < DSTATE; ++s) {
            h[s] = __expf(dtv * A[s]) * h[s] + du * sB[s];
            yv  += h[s] * sC[s];
        }
        yv += uv * Dl;
        float z = xz[row * 512 + DINNER + d];
        yv *= z / (1.f + __expf(-z));           // y * silu(z)
        dty[row * DINNER + d] = yv;
        __syncthreads();
    }
}

// --------------------------- final projection ------------------------------
__global__ void out_kernel(const float* __restrict__ fin,   // (M,128)
                           const float* __restrict__ outW,  // (7,128)
                           float* __restrict__ out) {
    int idx = blockIdx.x * blockDim.x + threadIdx.x;
    if (idx >= MROWS * COUT) return;
    int c   = idx % COUT;
    int row = idx / COUT;
    const float* x = fin  + (size_t)row * DMODEL;
    const float* w = outW + (size_t)c * DMODEL;
    float acc = 0.f;
#pragma unroll 16
    for (int d = 0; d < DMODEL; ++d) acc += x[d] * w[d];
    out[idx] = acc;
}

// ----------------------------- launcher ------------------------------------
extern "C" void kernel_launch(void* const* d_in, const int* in_sizes, int n_in,
                              void* d_out, int out_size, void* d_ws, size_t ws_size,
                              hipStream_t stream) {
    (void)in_sizes; (void)n_in; (void)out_size; (void)ws_size;
    const float* x_enc      = (const float*)d_in[0];
    const float* x_mark     = (const float*)d_in[1];
    // d_in[2], d_in[3] (x_dec, x_mark_dec) unused by the reference math
    const float* token_W    = (const float*)d_in[4];
    const float* temporal_W = (const float*)d_in[5];
    const float* norm_w     = (const float*)d_in[6];
    const float* norm_b     = (const float*)d_in[7];
    const float* in_proj_W  = (const float*)d_in[8];
    const float* conv_W     = (const float*)d_in[9];
    const float* conv_b     = (const float*)d_in[10];
    const float* x_proj_W   = (const float*)d_in[11];
    const float* dt_proj_W  = (const float*)d_in[12];
    const float* dt_proj_b  = (const float*)d_in[13];
    const float* A_log      = (const float*)d_in[14];
    const float* D_param    = (const float*)d_in[15];
    const float* out_proj_W = (const float*)d_in[16];
    const float* normf_w    = (const float*)d_in[17];
    const float* normf_b    = (const float*)d_in[18];
    const float* out_W      = (const float*)d_in[19];
    float* out = (float*)d_out;

    // workspace layout (floats); every buffer is written before it is read
    float* ws     = (float*)d_ws;
    const size_t M = MROWS;
    float* hidden = ws;                       // M*128
    float* resid  = hidden + M * DMODEL;      // M*128
    float* ln     = resid  + M * DMODEL;      // M*128
    float* xz     = ln     + M * DMODEL;      // M*512
    float* u      = xz     + M * 512;         // M*256
    float* dbc    = u      + M * DINNER;      // M*40
    float* dty    = dbc    + M * 40;          // M*256 (dt, then y in place)

    embed_kernel<<<MROWS, DMODEL, 0, stream>>>(x_enc, x_mark, token_W, temporal_W, hidden);

    const int tilesM = MROWS / 16;
    for (int l = 0; l < NLAYER; ++l) {
        ln_kernel<<<MROWS, DMODEL, 0, stream>>>(
            hidden, (l == 0) ? nullptr : resid, resid,
            norm_w + (size_t)l * DMODEL, norm_b + (size_t)l * DMODEL, ln);

        {   // in_proj: (M,128) x (512,128)^T -> xz (M,512)
            int tilesN2 = 512 / 32;
            int tiles   = tilesM * tilesN2;
            gemm_bf16_wmma<DMODEL><<<(tiles + 7) / 8, 256, 0, stream>>>(
                ln, in_proj_W + (size_t)l * 512 * DMODEL, xz, 512, tilesN2);
        }

        conv_silu_kernel<<<(MROWS * DINNER + 255) / 256, 256, 0, stream>>>(
            xz, conv_W + (size_t)l * DINNER * DCONV, conv_b + (size_t)l * DINNER, u);

        {   // x_proj: (M,256) x (40,256)^T -> dbc (M,40); padded to 2x32 tiles
            int tilesN2 = 2;
            int tiles   = tilesM * tilesN2;
            gemm_bf16_wmma<DINNER><<<(tiles + 7) / 8, 256, 0, stream>>>(
                u, x_proj_W + (size_t)l * 40 * DINNER, dbc, 40, tilesN2);
        }

        dt_kernel<<<(MROWS * DINNER + 255) / 256, 256, 0, stream>>>(
            dbc, dt_proj_W + (size_t)l * DINNER * DTRANK,
            dt_proj_b + (size_t)l * DINNER, dty);

        scan_kernel<<<BATCH, DINNER, 0, stream>>>(
            dbc, u, xz, A_log + (size_t)l * DINNER * DSTATE,
            D_param + (size_t)l * DINNER, dty);

        {   // out_proj: (M,256) x (128,256)^T -> hidden (M,128)
            int tilesN2 = DMODEL / 32;
            int tiles   = tilesM * tilesN2;
            gemm_bf16_wmma<DINNER><<<(tiles + 7) / 8, 256, 0, stream>>>(
                dty, out_proj_W + (size_t)l * DMODEL * DINNER, hidden, DMODEL, tilesN2);
        }
    }

    // final LN over (hidden + resid), then output projection
    ln_kernel<<<MROWS, DMODEL, 0, stream>>>(hidden, resid, nullptr, normf_w, normf_b, ln);
    out_kernel<<<(MROWS * COUT + 255) / 256, 256, 0, stream>>>(ln, out_W, out);
}
